// LegacyVMambaBlock_15393162788920
// MI455X (gfx1250) — compile-verified, matrix operand order checked by reading, not run on hardware
//
#include <hip/hip_runtime.h>

// ---------------------------------------------------------------------------
// LegacyVMambaBlock for MI455X (gfx1250, wave32, WMMA).
// All GEMMs: bf16 inputs, f32 accumulate via v_wmma_f32_16x16x32_bf16.
// Wave tile 32x64 (8 wmma / 6 fragment loads per K-step), block tile 64x128.
// ---------------------------------------------------------------------------

typedef __bf16 bf16;
typedef __attribute__((ext_vector_type(16))) __bf16 v16bf;
typedef __attribute__((ext_vector_type(8)))  __bf16 v8bf;
typedef __attribute__((ext_vector_type(8)))  float  v8f;

static constexpr int Bz = 16, Hh = 64, Wd = 64, Dd = 384;
static constexpr int INNER = 768, HIDDEN = 1536;
static constexpr int NTOK = Bz * Hh * Wd; // 65536
static constexpr float DECAY = 0.6f, GAIN = 0.4f;

// ---- fragment load: two contiguous 16B chunks per lane ---------------------
// 16-bit A 16x32 layout (ISA 7.12.2): lane hi-half selects K sub-block.
__device__ __forceinline__ v16bf load_frag(const bf16* rowptr, int k0, int hi) {
  v8bf lo = *reinterpret_cast<const v8bf*>(rowptr + k0 + hi * 8);
  v8bf hc = *reinterpret_cast<const v8bf*>(rowptr + k0 + 16 + hi * 8);
  v16bf f;
#pragma unroll
  for (int i = 0; i < 8; ++i) { f[i] = lo[i]; f[8 + i] = hc[i]; }
  return f;
}

__device__ __forceinline__ float gelu_exact(float x) {
  return 0.5f * x * (1.0f + erff(x * 0.70710678118654752f));
}

// ---- weight cast + transpose: src f32 [K][N] -> dst bf16 [N][K] -----------
__global__ void castT_kernel(const float* __restrict__ src, bf16* __restrict__ dst,
                             int K, int N) {
  size_t idx = (size_t)blockIdx.x * blockDim.x + threadIdx.x;
  if (idx >= (size_t)K * N) return;
  int n = (int)(idx % N);
  int k = (int)(idx / N);
  dst[(size_t)n * K + k] = (bf16)src[idx];
}

// ---- LayerNorm (+ optional 2-input combine) -> bf16, one wave32 per token --
template <int D>
__global__ void ln_cast_kernel(const float* __restrict__ in1, const float* __restrict__ in2,
                               float s1, const float* __restrict__ g,
                               const float* __restrict__ b, bf16* __restrict__ out) {
  constexpr int PER = D / 32;
  int lane = threadIdx.x & 31;
  int wave = threadIdx.x >> 5;
  int token = blockIdx.x * (blockDim.x >> 5) + wave;
  size_t base = (size_t)token * D;
  float vals[PER];
  float sum = 0.f, sumsq = 0.f;
#pragma unroll
  for (int i = 0; i < PER; ++i) {
    int j = lane + i * 32; // coalesced
    float v = in1[base + j] * s1;
    if (in2) v += in2[base + j];
    vals[i] = v;
    sum += v;
    sumsq += v * v;
  }
#pragma unroll
  for (int off = 16; off; off >>= 1) {
    sum += __shfl_xor(sum, off, 32);
    sumsq += __shfl_xor(sumsq, off, 32);
  }
  float mean = sum * (1.0f / D);
  float var = sumsq * (1.0f / D) - mean * mean;
  float rstd = rsqrtf(var + 1e-5f);
#pragma unroll
  for (int i = 0; i < PER; ++i) {
    int j = lane + i * 32;
    out[base + j] = (bf16)((vals[i] - mean) * rstd * g[j] + b[j]);
  }
}

// ---- bf16 WMMA GEMM: out[M,N] = act(A[M,K] @ Bt[N,K]^T + bias) (+resid) ----
// block = 128 thr (4 waves, 2x2), block tile 64x128, wave tile 32x64.
// Per K-step: 2 A-frags + 4 B-frags feed 8 independent wmma chains.
template <int ACT, int RESID, int OUTBF>
__global__ void gemm_kernel(const bf16* __restrict__ A, const bf16* __restrict__ Bt,
                            const float* __restrict__ bias, const float* __restrict__ resid,
                            float* __restrict__ outF, bf16* __restrict__ outB,
                            int M, int N, int K) {
  int l = threadIdx.x & 31;
  int wave = threadIdx.x >> 5;
  int wm = (wave >> 1) * 32, wn = (wave & 1) * 64;
  int rm = blockIdx.x * 64 + wm;
  int cn = blockIdx.y * 128 + wn;
  int mrow = l & 15, hi = l >> 4;

  v8f c00 = {}, c01 = {}, c02 = {}, c03 = {};
  v8f c10 = {}, c11 = {}, c12 = {}, c13 = {};
  const bf16* a0p = A + (size_t)(rm + mrow) * K;
  const bf16* a1p = A + (size_t)(rm + 16 + mrow) * K;
  const bf16* b0p = Bt + (size_t)(cn + mrow) * K;
  const bf16* b1p = Bt + (size_t)(cn + 16 + mrow) * K;
  const bf16* b2p = Bt + (size_t)(cn + 32 + mrow) * K;
  const bf16* b3p = Bt + (size_t)(cn + 48 + mrow) * K;

  for (int kk = 0; kk < K; kk += 32) {
    v16bf a0 = load_frag(a0p, kk, hi);
    v16bf a1 = load_frag(a1p, kk, hi);
    v16bf b0 = load_frag(b0p, kk, hi);
    v16bf b1 = load_frag(b1p, kk, hi);
    v16bf b2 = load_frag(b2p, kk, hi);
    v16bf b3 = load_frag(b3p, kk, hi);
    c00 = __builtin_amdgcn_wmma_f32_16x16x32_bf16(false, a0, false, b0, (short)0, c00, false, false);
    c01 = __builtin_amdgcn_wmma_f32_16x16x32_bf16(false, a0, false, b1, (short)0, c01, false, false);
    c02 = __builtin_amdgcn_wmma_f32_16x16x32_bf16(false, a0, false, b2, (short)0, c02, false, false);
    c03 = __builtin_amdgcn_wmma_f32_16x16x32_bf16(false, a0, false, b3, (short)0, c03, false, false);
    c10 = __builtin_amdgcn_wmma_f32_16x16x32_bf16(false, a1, false, b0, (short)0, c10, false, false);
    c11 = __builtin_amdgcn_wmma_f32_16x16x32_bf16(false, a1, false, b1, (short)0, c11, false, false);
    c12 = __builtin_amdgcn_wmma_f32_16x16x32_bf16(false, a1, false, b2, (short)0, c12, false, false);
    c13 = __builtin_amdgcn_wmma_f32_16x16x32_bf16(false, a1, false, b3, (short)0, c13, false, false);
  }

  // C/D layout (ISA 7.12.2): VGPR v -> row v + 8*hi, col = lane&15.
  v8f* fr[8] = {&c00, &c01, &c02, &c03, &c10, &c11, &c12, &c13};
#pragma unroll
  for (int f = 0; f < 8; ++f) {
    int mi = f >> 2, ni = f & 3;
    int col = cn + ni * 16 + mrow;
    float bv = bias[col];
    v8f& cc = *fr[f];
#pragma unroll
    for (int v = 0; v < 8; ++v) {
      int row = rm + mi * 16 + v + 8 * hi;
      size_t idx = (size_t)row * N + col;
      float val = cc[v] + bv;
      if (ACT == 1) val = gelu_exact(val);
      if (RESID) val += resid[idx];
      if (OUTBF) outB[idx] = (bf16)val;
      else       outF[idx] = val;
    }
  }
}

// ---- depthwise 3x3 SAME conv over (H,W), channel-fastest layout ------------
__global__ void dwconv_kernel(const float* __restrict__ p, const float* __restrict__ kc,
                              float* __restrict__ local) {
  size_t idx = (size_t)blockIdx.x * blockDim.x + threadIdx.x;
  if (idx >= (size_t)NTOK * INNER) return;
  int c = (int)(idx % INNER);
  size_t t = idx / INNER;
  int w = (int)(t % Wd); t /= Wd;
  int h = (int)(t % Hh);
  int b = (int)(t / Hh);
  float acc = 0.f;
#pragma unroll
  for (int dy = -1; dy <= 1; ++dy) {
#pragma unroll
    for (int dx = -1; dx <= 1; ++dx) {
      int hh = h + dy, ww = w + dx;
      if (hh >= 0 && hh < Hh && ww >= 0 && ww < Wd) {
        size_t pidx = ((((size_t)b * Hh + hh) * Wd + ww) * INNER) + c;
        acc += p[pidx] * kc[((dy + 1) * 3 + (dx + 1)) * INNER + c];
      }
    }
  }
  local[idx] = acc;
}

// ---- EMA scans along W (fwd+rev), acc = lr + rl ----------------------------
__global__ void scanW_kernel(const float* __restrict__ local, float* __restrict__ acc) {
  size_t idx = (size_t)blockIdx.x * blockDim.x + threadIdx.x;
  if (idx >= (size_t)Bz * Hh * INNER) return;
  int c = (int)(idx % INNER);
  size_t t = idx / INNER;
  int h = (int)(t % Hh);
  int b = (int)(t / Hh);
  size_t base = (((size_t)b * Hh + h) * Wd) * INNER + c;
  float s = 0.f;
  for (int w = 0; w < Wd; ++w) {
    s = DECAY * s + GAIN * local[base + (size_t)w * INNER];
    acc[base + (size_t)w * INNER] = s;
  }
  s = 0.f;
  for (int w = Wd - 1; w >= 0; --w) {
    s = DECAY * s + GAIN * local[base + (size_t)w * INNER];
    acc[base + (size_t)w * INNER] += s;
  }
}

// ---- EMA scans along H (fwd+rev), acc += tb + bt ---------------------------
__global__ void scanH_kernel(const float* __restrict__ local, float* __restrict__ acc) {
  size_t idx = (size_t)blockIdx.x * blockDim.x + threadIdx.x;
  if (idx >= (size_t)Bz * Wd * INNER) return;
  int c = (int)(idx % INNER);
  size_t t = idx / INNER;
  int w = (int)(t % Wd);
  int b = (int)(t / Wd);
  size_t base = (((size_t)b * Hh) * Wd + w) * INNER + c;
  size_t stride = (size_t)Wd * INNER;
  float s = 0.f;
  for (int h = 0; h < Hh; ++h) {
    s = DECAY * s + GAIN * local[base + (size_t)h * stride];
    acc[base + (size_t)h * stride] += s;
  }
  s = 0.f;
  for (int h = Hh - 1; h >= 0; --h) {
    s = DECAY * s + GAIN * local[base + (size_t)h * stride];
    acc[base + (size_t)h * stride] += s;
  }
}

// ---------------------------------------------------------------------------
extern "C" void kernel_launch(void* const* d_in, const int* in_sizes, int n_in,
                              void* d_out, int out_size, void* d_ws, size_t ws_size,
                              hipStream_t stream) {
  const float* x      = (const float*)d_in[0];
  const float* n1_g   = (const float*)d_in[1];
  const float* n1_b   = (const float*)d_in[2];
  const float* w_in   = (const float*)d_in[3];
  const float* b_in   = (const float*)d_in[4];
  const float* k_conv = (const float*)d_in[5];
  const float* on_g   = (const float*)d_in[6];
  const float* on_b   = (const float*)d_in[7];
  const float* w_out  = (const float*)d_in[8];
  const float* b_out  = (const float*)d_in[9];
  const float* n2_g   = (const float*)d_in[10];
  const float* n2_b   = (const float*)d_in[11];
  const float* w1     = (const float*)d_in[12];
  const float* b1     = (const float*)d_in[13];
  const float* w2     = (const float*)d_in[14];
  const float* b2     = (const float*)d_in[15];
  float* out = (float*)d_out;

  // ---- workspace layout (256B aligned), ~531 MB with aggressive reuse ------
  char* wsb = (char*)d_ws;
  size_t off = 0;
  auto alloc = [&](size_t bytes) {
    void* p = wsb + off;
    off = (off + bytes + 255) & ~(size_t)255;
    return p;
  };
  bf16* wInT   = (bf16*)alloc((size_t)Dd * INNER * 2);       // [768][384]
  bf16* wOutT  = (bf16*)alloc((size_t)INNER * Dd * 2);       // [384][768]
  bf16* w1T    = (bf16*)alloc((size_t)Dd * HIDDEN * 2);      // [1536][384]
  bf16* w2T    = (bf16*)alloc((size_t)HIDDEN * Dd * 2);      // [384][1536]
  bf16* hb     = (bf16*)alloc((size_t)NTOK * Dd * 2);        // LN1(x); reused as h2b
  float* p     = (float*)alloc((size_t)NTOK * INNER * 4);    // GEMM1 out; reused as acc
  float* local = (float*)alloc((size_t)NTOK * INNER * 4);    // conv out; reused as gb (bf16)
  bf16* mixedb = (bf16*)alloc((size_t)NTOK * INNER * 2);     // LN(mixed) bf16
  float* acc = p;
  bf16* h2b = hb;
  bf16* gb = (bf16*)local; // 65536*1536*2 == 65536*768*4 bytes exactly

  // ---- 1. weight cast+transpose to bf16 ------------------------------------
  {
    int t = 256;
    castT_kernel<<<(Dd * INNER + t - 1) / t, t, 0, stream>>>(w_in, wInT, Dd, INNER);
    castT_kernel<<<(INNER * Dd + t - 1) / t, t, 0, stream>>>(w_out, wOutT, INNER, Dd);
    castT_kernel<<<(Dd * HIDDEN + t - 1) / t, t, 0, stream>>>(w1, w1T, Dd, HIDDEN);
    castT_kernel<<<(HIDDEN * Dd + t - 1) / t, t, 0, stream>>>(w2, w2T, HIDDEN, Dd);
  }

  // ---- 2. LN1(x) -> hb (bf16) ----------------------------------------------
  ln_cast_kernel<Dd><<<NTOK / 4, 128, 0, stream>>>(x, nullptr, 1.0f, n1_g, n1_b, hb);

  // ---- 3. GEMM1: p = hb @ w_in + b_in  (f32 out) ---------------------------
  gemm_kernel<0, 0, 0><<<dim3(NTOK / 64, INNER / 128), 128, 0, stream>>>(
      hb, wInT, b_in, nullptr, p, nullptr, NTOK, INNER, Dd);

  // ---- 4. depthwise 3x3 conv: local = conv(p) ------------------------------
  {
    size_t total = (size_t)NTOK * INNER;
    dwconv_kernel<<<(unsigned)((total + 255) / 256), 256, 0, stream>>>(p, k_conv, local);
  }

  // ---- 5. EMA scans: acc = lr + rl + tb + bt (acc aliases p) ---------------
  {
    size_t nW = (size_t)Bz * Hh * INNER;
    scanW_kernel<<<(unsigned)((nW + 255) / 256), 256, 0, stream>>>(local, acc);
    size_t nH = (size_t)Bz * Wd * INNER;
    scanH_kernel<<<(unsigned)((nH + 255) / 256), 256, 0, stream>>>(local, acc);
  }

  // ---- 6. mixed = LN(0.25*acc + local) -> mixedb (bf16) --------------------
  ln_cast_kernel<INNER><<<NTOK / 4, 128, 0, stream>>>(acc, local, 0.25f, on_g, on_b, mixedb);

  // ---- 7. GEMM2: out = x + mixedb @ w_out + b_out --------------------------
  gemm_kernel<0, 1, 0><<<dim3(NTOK / 64, Dd / 128), 128, 0, stream>>>(
      mixedb, wOutT, b_out, x, out, nullptr, NTOK, Dd, INNER);

  // ---- 8. LN2(out) -> h2b (bf16, reuses hb) --------------------------------
  ln_cast_kernel<Dd><<<NTOK / 4, 128, 0, stream>>>(out, nullptr, 1.0f, n2_g, n2_b, h2b);

  // ---- 9. MLP1: gb = gelu(h2b @ w1 + b1) (bf16, reuses local) --------------
  gemm_kernel<1, 0, 1><<<dim3(NTOK / 64, HIDDEN / 128), 128, 0, stream>>>(
      h2b, w1T, b1, nullptr, nullptr, gb, NTOK, HIDDEN, Dd);

  // ---- 10. MLP2: out = out + gb @ w2 + b2 ----------------------------------
  gemm_kernel<0, 1, 0><<<dim3(NTOK / 64, Dd / 128), 128, 0, stream>>>(
      gb, w2T, b2, out, out, nullptr, NTOK, Dd, HIDDEN);

  (void)in_sizes; (void)n_in; (void)out_size; (void)ws_size;
}
